// VariableSelectionNetwork_62723702391471
// MI455X (gfx1250) — compile-verified, matrix operand order checked by reading, not run on hardware
//
#include <hip/hip_runtime.h>
#include <hip/hip_bf16.h>

typedef __attribute__((ext_vector_type(16))) _Float16 v16h;
typedef __attribute__((ext_vector_type(8)))  float    v8f;

#define LN_EPS 1e-5f

__device__ __forceinline__ float elu_f(float z)     { return z > 0.f ? z : __expf(z) - 1.f; }
__device__ __forceinline__ float sigmoid_f(float z) { return 1.f / (1.f + __expf(-z)); }

// ---------------------------------------------------------------------------
// Kernel 1: weight GRN + softmax over F=32.  One wave32 per (b,s) position.
// lane j owns feature j.
// ---------------------------------------------------------------------------
__global__ __launch_bounds__(256) void vsn_weights_kernel(
    const float* __restrict__ x,     // [npos,32]
    const float* __restrict__ w1,    // [32,32]
    const float* __restrict__ b1,    // [32]
    const float* __restrict__ w2,    // [32,64]
    const float* __restrict__ b2,    // [64]
    const float* __restrict__ lng,   // [32]
    const float* __restrict__ lnb,   // [32]
    float* __restrict__ wout,        // [npos,32]
    int npos)
{
    int wid  = (blockIdx.x * blockDim.x + threadIdx.x) >> 5;
    int lane = threadIdx.x & 31;
    if (wid >= npos) return;

    float xj = x[wid * 32 + lane];

    // fc1 + ELU : h[j] = elu(sum_i x_i * w1[i][j] + b1[j])
    float acc = b1[lane];
    #pragma unroll 8
    for (int i = 0; i < 32; ++i)
        acc += __shfl(xj, i, 32) * w1[i * 32 + lane];
    float h = elu_f(acc);

    // fc2 : lane j holds h2[j] and h2[j+32]
    float a0 = b2[lane], a1 = b2[lane + 32];
    #pragma unroll 8
    for (int i = 0; i < 32; ++i) {
        float hi = __shfl(h, i, 32);
        a0 += hi * w2[i * 64 + lane];
        a1 += hi * w2[i * 64 + lane + 32];
    }

    // GLU + identity residual
    float y = xj + a0 * sigmoid_f(a1);

    // LayerNorm over the 32 lanes
    float s = y, ss = y * y;
    #pragma unroll
    for (int m = 16; m; m >>= 1) { s += __shfl_xor(s, m, 32); ss += __shfl_xor(ss, m, 32); }
    float mean = s * (1.f / 32.f);
    float var  = ss * (1.f / 32.f) - mean * mean;
    float z = (y - mean) * rsqrtf(var + LN_EPS) * lng[lane] + lnb[lane];

    // softmax over 32 lanes
    float mx = z;
    #pragma unroll
    for (int m = 16; m; m >>= 1) mx = fmaxf(mx, __shfl_xor(mx, m, 32));
    float e = __expf(z - mx);
    float es = e;
    #pragma unroll
    for (int m = 16; m; m >>= 1) es += __shfl_xor(es, m, 32);

    wout[wid * 32 + lane] = e / es;
}

// ---------------------------------------------------------------------------
// Kernel 2: fused feature GRN + weighted combine.
// Per (b,s): [32 x 128] @ [128 x 256] WMMA GEMM (A = elu(x*w1+b1) in f16,
// B = w2 in f16 pre-swizzled into LDS), GLU+LN+weighted-sum in registers/LDS.
//
// WMMA 16-bit A layout (16x32 MxK), lane l, vector element e (0..15):
//   row  m = l & 15
//   k(e,l) = (e < 8 ? e : e + 8) + ((l >= 16) ? 8 : 0)
// B (32x16 KxN) uses the transposed-symmetric layout: col = l & 15, same k(e,l).
// C/D f32 16x16: VGPR r, lanes 0-15 -> row r, lanes 16-31 -> row r+8; col = l&15.
// ---------------------------------------------------------------------------
__global__ __launch_bounds__(256) void vsn_main_kernel(
    const float* __restrict__ x,      // [npos,32]
    const float* __restrict__ wts,    // [npos,32] softmax weights
    const float* __restrict__ fw1,    // [128]
    const float* __restrict__ fb1,    // [128]
    const float* __restrict__ fw2,    // [128,256]
    const float* __restrict__ fb2,    // [256]
    const float* __restrict__ fsw,    // [128]
    const float* __restrict__ fsb,    // [128]
    const float* __restrict__ flng,   // [128]
    const float* __restrict__ flnb,   // [128]
    float* __restrict__ out,          // [npos,128]
    int npos)
{
    // ---- LDS (static; ~78.8 KB -> 4 WGs per 320KB WGP) ----
    __shared__ __align__(32) _Float16 sB[16 * 4 * 32 * 16]; // 64 KB: w2 as WMMA-B frags [nt][ks][lane][e]
    __shared__ __align__(32) _Float16 sA[8 * 32 * 16];      // 8 KB : A frags [mt*4+ks][lane][e]
    __shared__ float sC[1024];   // 0:w1 128:b1 256:b2(256) 512:sw 640:sb 768:lng 896:lnb
    __shared__ float sX[32];
    __shared__ float sWt[32];
    __shared__ float sStat[64];  // [row][2] = {sum, sumsq}
    __shared__ float sOut[128];

    const int t    = threadIdx.x;
    const int lane = t & 31;
    const int wave = t >> 5;
    const int mt   = wave >> 2;       // M-tile (rows f 0-15 / 16-31)
    const int q    = wave & 3;        // N-tile pair group
    const int half = lane >> 4;       // 0: C rows r, 1: C rows r+8
    const int ln15 = lane & 15;

    // ---- one-time setup: constants + w2 -> f16 WMMA-B fragments ----
    for (int i = t; i < 1024; i += 256) {
        float v;
        if      (i < 128) v = fw1[i];
        else if (i < 256) v = fb1[i - 128];
        else if (i < 512) v = fb2[i - 256];
        else if (i < 640) v = fsw[i - 512];
        else if (i < 768) v = fsb[i - 640];
        else if (i < 896) v = flng[i - 768];
        else              v = flnb[i - 896];
        sC[i] = v;
    }
    for (int idx = t; idx < 16 * 4 * 32 * 16; idx += 256) {
        int e  = idx & 15;
        int l  = (idx >> 4) & 31;
        int ks = (idx >> 9) & 3;
        int nt = idx >> 11;
        int k  = ks * 32 + (e < 8 ? e : e + 8) + ((l >= 16) ? 8 : 0);
        int n  = nt * 16 + (l & 15);
        sB[idx] = (_Float16)fw2[k * 256 + n];
    }
    __syncthreads();

    const int nt0 = 2 * q;          // GLU value tiles: nt0, nt0+1 ; gates: +8
    const int nt1 = 2 * q + 1;
    const int j0  = nt0 * 16 + ln15;   // output columns owned by this lane
    const int j1  = nt1 * 16 + ln15;
    const float b2v0 = sC[256 + j0],       b2v1 = sC[256 + j1];
    const float b2g0 = sC[256 + 128 + j0], b2g1 = sC[256 + 128 + j1];
    const float sw0  = sC[512 + j0], sw1 = sC[512 + j1];
    const float sb0  = sC[640 + j0], sb1 = sC[640 + j1];
    const float g0c  = sC[768 + j0], g1c = sC[768 + j1];
    const float bb0  = sC[896 + j0], bb1 = sC[896 + j1];

    for (int pos = blockIdx.x; pos < npos; pos += gridDim.x) {
        // ---- phase 1: zero accumulators, stage x/weights, build A frags ----
        if (t < 128)              sOut[t] = 0.f;
        else if (t < 192)         sStat[t - 128] = 0.f;
        if (t < 32)               sX[t] = x[(size_t)pos * 32 + t];
        else if (t >= 32 && t < 64) sWt[t - 32] = wts[(size_t)pos * 32 + (t - 32)];
        {
            // wave == fragment index (mt = wave>>2, ks = wave&3)
            int f  = (mt << 4) + ln15;
            float xf = x[(size_t)pos * 32 + f];
            int kb = q * 32 + (half ? 8 : 0);
            _Float16* dst = sA + ((size_t)wave * 32 + lane) * 16;
            #pragma unroll
            for (int e = 0; e < 16; ++e) {
                int k = kb + (e < 8 ? e : e + 8);
                float hv = fmaf(xf, sC[k], sC[128 + k]);
                dst[e] = (_Float16)elu_f(hv);
            }
        }
        __syncthreads();

        // ---- phase 2: WMMA GEMM, K = 128 in 4 steps of 32 ----
        v8f accV0 = {}, accG0 = {}, accV1 = {}, accG1 = {};
        #pragma unroll
        for (int ks = 0; ks < 4; ++ks) {
            v16h a   = *(const v16h*)(sA + ((mt * 4 + ks) * 32 + lane) * 16);
            v16h bv0 = *(const v16h*)(sB + (((nt0    ) * 4 + ks) * 32 + lane) * 16);
            v16h bg0 = *(const v16h*)(sB + (((nt0 + 8) * 4 + ks) * 32 + lane) * 16);
            v16h bv1 = *(const v16h*)(sB + (((nt1    ) * 4 + ks) * 32 + lane) * 16);
            v16h bg1 = *(const v16h*)(sB + (((nt1 + 8) * 4 + ks) * 32 + lane) * 16);
            accV0 = __builtin_amdgcn_wmma_f32_16x16x32_f16(false, a, false, bv0, (short)0, accV0, false, false);
            accG0 = __builtin_amdgcn_wmma_f32_16x16x32_f16(false, a, false, bg0, (short)0, accG0, false, false);
            accV1 = __builtin_amdgcn_wmma_f32_16x16x32_f16(false, a, false, bv1, (short)0, accV1, false, false);
            accG1 = __builtin_amdgcn_wmma_f32_16x16x32_f16(false, a, false, bg1, (short)0, accG1, false, false);
        }

        // ---- GLU + residual, LN row partials (sum / sumsq over 128 cols) ----
        float y0[8], y1[8];
        #pragma unroll
        for (int r = 0; r < 8; ++r) {
            int f = mt * 16 + r + half * 8;
            float xf = sX[f];
            float v0 = (accV0[r] + b2v0) * sigmoid_f(accG0[r] + b2g0);
            float v1 = (accV1[r] + b2v1) * sigmoid_f(accG1[r] + b2g1);
            y0[r] = fmaf(xf, sw0, sb0) + v0;
            y1[r] = fmaf(xf, sw1, sb1) + v1;
            float s  = y0[r] + y1[r];
            float ss = y0[r] * y0[r] + y1[r] * y1[r];
            #pragma unroll
            for (int m = 1; m <= 8; m <<= 1) {      // reduce within each 16-lane half
                s  += __shfl_xor(s,  m, 32);
                ss += __shfl_xor(ss, m, 32);
            }
            if (ln15 == 0) {                        // lanes 0 and 16 commit row partials
                atomicAdd(&sStat[f * 2 + 0], s);
                atomicAdd(&sStat[f * 2 + 1], ss);
            }
        }
        __syncthreads();

        // ---- phase 3: normalize, scale by softmax weight, reduce over F ----
        float oa0 = 0.f, oa1 = 0.f;
        #pragma unroll
        for (int r = 0; r < 8; ++r) {
            int f = mt * 16 + r + half * 8;
            float mean = sStat[f * 2 + 0] * (1.f / 128.f);
            float var  = sStat[f * 2 + 1] * (1.f / 128.f) - mean * mean;
            float inv  = rsqrtf(var + LN_EPS);
            float wf   = sWt[f];
            oa0 += wf * (fmaf((y0[r] - mean) * inv, g0c, bb0));
            oa1 += wf * (fmaf((y1[r] - mean) * inv, g1c, bb1));
        }
        oa0 += __shfl_xor(oa0, 16, 32);             // fold lane halves (same column)
        oa1 += __shfl_xor(oa1, 16, 32);
        if (half == 0) {                            // mt=0 and mt=1 waves both add
            atomicAdd(&sOut[j0], oa0);
            atomicAdd(&sOut[j1], oa1);
        }
        __syncthreads();

        // ---- phase 4: store ----
        if (t < 128) out[(size_t)pos * 128 + t] = sOut[t];
        __syncthreads();
    }
}

extern "C" void kernel_launch(void* const* d_in, const int* in_sizes, int n_in,
                              void* d_out, int out_size, void* d_ws, size_t ws_size,
                              hipStream_t stream) {
    const float* x      = (const float*)d_in[0];
    const float* wg_w1  = (const float*)d_in[1];
    const float* wg_b1  = (const float*)d_in[2];
    const float* wg_w2  = (const float*)d_in[3];
    const float* wg_b2  = (const float*)d_in[4];
    const float* wg_lng = (const float*)d_in[5];
    const float* wg_lnb = (const float*)d_in[6];
    const float* fg_w1  = (const float*)d_in[7];
    const float* fg_b1  = (const float*)d_in[8];
    const float* fg_w2  = (const float*)d_in[9];
    const float* fg_b2  = (const float*)d_in[10];
    const float* fg_sw  = (const float*)d_in[11];
    const float* fg_sb  = (const float*)d_in[12];
    const float* fg_lng = (const float*)d_in[13];
    const float* fg_lnb = (const float*)d_in[14];

    const int npos = in_sizes[0] / 32;      // B*S
    float* wts = (float*)d_ws;              // [npos,32] softmax weights

    vsn_weights_kernel<<<(npos + 7) / 8, 256, 0, stream>>>(
        x, wg_w1, wg_b1, wg_w2, wg_b2, wg_lng, wg_lnb, wts, npos);

    int grid = npos < 2048 ? npos : 2048;   // 8 positions per WG at npos=16384
    vsn_main_kernel<<<grid, 256, 0, stream>>>(
        x, wts, fg_w1, fg_b1, fg_w2, fg_b2, fg_sw, fg_sb, fg_lng, fg_lnb,
        (float*)d_out, npos);
}
